// TextGenerationModel_10780367913063
// MI455X (gfx1250) — matrix operand order repaired and checked by reading, not compiled, not run on hardware
//
#include <hip/hip_runtime.h>
#include <hip/hip_bf16.h>
#include <stdint.h>

// ---------------------------------------------------------------------------
// 2-layer LSTM LM on MI455X (gfx1250, wave32, WMMA + TDM).
//
//  - bf16 WMMA (v_wmma_f32_16x16x32_bf16) for all GEMMs, fp32 accumulate.
//  - Batch GEMMs: 64x64 tile / 4-wave block, LDS double-buffered k-loop.
//  - Recurrent scan: persistent 16-WG kernel, one grid barrier per timestep.
//    W_hh tile loaded once into LDS via TDM (3D descriptor: 256 x 16 rows x
//    4 gates); per-step h[64,256] staged via a 2D TDM tensor_load_to_lds;
//    c-state in registers; each wave computes i,f,g,o accumulators for the
//    same 16x16 (batch x h-col) tile so the cell update is lane-local.
// ---------------------------------------------------------------------------

typedef __attribute__((ext_vector_type(16))) __bf16   v16bf;
typedef __attribute__((ext_vector_type(8)))  float    v8f;
typedef __attribute__((ext_vector_type(4)))  unsigned u32x4;
typedef __attribute__((ext_vector_type(8)))  unsigned u32x8;

constexpr int kB  = 64;
constexpr int kT  = 1024;
constexpr int kV  = 256;
constexpr int kH  = 256;
constexpr int kH4 = 1024;
constexpr int kBT = kB * kT;
constexpr int NWG = 16;          // workgroups in the persistent scan kernel

// ---------------- fragment builders (ISA 7.12.2 layouts, wave32) -----------

// 16-bit A-matrix 16x32: lane = 16*half + row; VGPR v holds K pair:
//   k = (v<4?0:16) + half*8 + (v%4)*2
__device__ inline v16bf frag_a(const __bf16* __restrict__ s, int stride,
                               int row0, int k0, int lane) {
  union { v16bf v; unsigned u[8]; } f;
  const int row  = row0 + (lane & 15);
  const int half = lane >> 4;
#pragma unroll
  for (int v = 0; v < 8; ++v) {
    const int k = k0 + ((v & 4) << 2) + half * 8 + ((v & 3) << 1);
    f.u[v] = *(const unsigned*)(s + row * stride + k);
  }
  return f.v;
}

// 16-bit B-matrix 32x16 (from row-major W[N][K]): lane = 16*half + n;
// VGPR v holds K pair: k = (v<4?0:16) + half*8 + (v%4)*2
__device__ inline v16bf frag_b(const __bf16* __restrict__ s, int stride,
                               int row0, int k0, int lane) {
  union { v16bf v; unsigned u[8]; } f;
  const int n    = row0 + (lane & 15);
  const int half = lane >> 4;
#pragma unroll
  for (int v = 0; v < 8; ++v) {
    const int k = k0 + ((v & 4) << 2) + half * 8 + ((v & 3) << 1);
    f.u[v] = *(const unsigned*)(s + n * stride + k);
  }
  return f.v;
}

__device__ inline v8f wmma_bf16(v16bf a, v16bf b, v8f c) {
  return __builtin_amdgcn_wmma_f32_16x16x32_bf16(false, a, false, b,
                                                 (short)0, c, false, false);
}

__device__ inline float sigmoidf_(float x) {
  return 1.0f / (1.0f + __expf(-x));
}

// ---------------- Tensor Data Mover (ISA ch.8 D# descriptors) --------------
// 2D tile load: dim0 elements (bf16) x dim1 rows, row stride stride0 elems.
// D# group0: count=1 | lds_addr | global_addr | type=2 ("image").
// D# group1: data_size=2B, tensor/tile dims, dim0 stride.

__device__ inline void tdm_load_2d_bf16(unsigned lds_off, const void* gptr,
                                        unsigned dim0, unsigned dim1,
                                        unsigned stride0) {
  const unsigned long long ga = (unsigned long long)(uintptr_t)gptr;
  const u32x4 g0 = { 1u, lds_off, (unsigned)ga,
                     (unsigned)((ga >> 32) & 0x1FFFFFFu) | (2u << 30) };
  const u32x8 g1 = { 1u << 16,                                // data_size=2B
                     (dim0 & 0xFFFFu) << 16,                  // tensor_dim0 lo
                     (dim0 >> 16) | ((dim1 & 0xFFFFu) << 16), // dim0 hi|dim1 lo
                     (dim1 >> 16) | ((dim0 & 0xFFFFu) << 16), // dim1 hi|tile_d0
                     dim1 & 0xFFFFu,                          // tile_dim1
                     stride0,                                 // dim0_stride lo
                     0u, 0u };
  asm volatile("tensor_load_to_lds %0, %1 scope:SCOPE_DEV"
               :: "s"(g0), "s"(g1) : "memory");
}

// 3D tile load: dim0 elems x rows (stride0) x planes (stride1, elems).
__device__ inline void tdm_load_3d_bf16(unsigned lds_off, const void* gptr,
                                        unsigned dim0, unsigned rows,
                                        unsigned planes, unsigned stride0,
                                        unsigned long long stride1,
                                        unsigned tensor_rows) {
  const unsigned long long ga = (unsigned long long)(uintptr_t)gptr;
  const u32x4 g0 = { 1u, lds_off, (unsigned)ga,
                     (unsigned)((ga >> 32) & 0x1FFFFFFu) | (2u << 30) };
  const u32x8 g1 = { 1u << 16,
                     (dim0 & 0xFFFFu) << 16,
                     (dim0 >> 16) | ((tensor_rows & 0xFFFFu) << 16),
                     (tensor_rows >> 16) | ((dim0 & 0xFFFFu) << 16),
                     (rows & 0xFFFFu) | ((planes & 0xFFFFu) << 16),
                     stride0,
                     (unsigned)((stride1 & 0xFFFFu) << 16),   // dim1_stride lo16
                     (unsigned)(stride1 >> 16) };             // dim1_stride hi
  const u32x4 g2 = { planes, 0u, 0u, 0u };                    // tensor_dim2
  const u32x4 g3 = { 0u, 0u, 0u, 0u };
  asm volatile("tensor_load_to_lds %0, %1, %2, %3"
               :: "s"(g0), "s"(g1), "s"(g2), "s"(g3) : "memory");
}

// ---------------- fp32 -> bf16 conversion ----------------------------------

__global__ void cvt_f32_bf16(const float* __restrict__ src,
                             __bf16* __restrict__ dst, int n) {
  for (int i = blockIdx.x * blockDim.x + threadIdx.x; i < n;
       i += gridDim.x * blockDim.x)
    dst[i] = (__bf16)src[i];
}

__global__ void add_vec(const float* __restrict__ a,
                        const float* __restrict__ b,
                        float* __restrict__ o, int n) {
  int i = blockIdx.x * blockDim.x + threadIdx.x;
  if (i < n) o[i] = a[i] + b[i];
}

// ---------------- tiled bf16 GEMM: C[M,N] = A[M,K] * W[N,K]^T + bias -------
// block = 128 threads (4 waves); tile 64x64; wave w owns rows w*16..w*16+15
// and all four 16-wide n-tiles. LDS ping-pong double buffer on the k-loop.

__global__ __launch_bounds__(128) void gemm_bf16_nt(
    const __bf16* __restrict__ A, const __bf16* __restrict__ W,
    const float* __restrict__ bias, float* __restrict__ C,
    int M, int N, int K) {
  __shared__ __bf16 A_s[2][64 * 32];
  __shared__ __bf16 B_s[2][64 * 32];

  const int m0   = blockIdx.y * 64;
  const int n0   = blockIdx.x * 64;
  const int tid  = threadIdx.x;
  const int lane = tid & 31;
  const int wid  = tid >> 5;
  const int row0 = tid >> 2;          // chunk rows 0..31
  const int row1 = row0 + 32;         // chunk rows 32..63
  const int c4   = (tid & 3) * 8;     // 8-elem column chunk

  v8f acc[4];
#pragma unroll
  for (int i = 0; i < 4; ++i) acc[i] = (v8f){0, 0, 0, 0, 0, 0, 0, 0};

  const int nk = K >> 5;

  // prologue: stage k-tile 0
  uint4 ra0 = *(const uint4*)&A[(size_t)(m0 + row0) * K + c4];
  uint4 ra1 = *(const uint4*)&A[(size_t)(m0 + row1) * K + c4];
  uint4 rb0 = *(const uint4*)&W[(size_t)(n0 + row0) * K + c4];
  uint4 rb1 = *(const uint4*)&W[(size_t)(n0 + row1) * K + c4];
  *(uint4*)&A_s[0][row0 * 32 + c4] = ra0;
  *(uint4*)&A_s[0][row1 * 32 + c4] = ra1;
  *(uint4*)&B_s[0][row0 * 32 + c4] = rb0;
  *(uint4*)&B_s[0][row1 * 32 + c4] = rb1;
  __syncthreads();

  for (int ki = 0; ki < nk; ++ki) {
    const int  p    = ki & 1;
    const bool more = (ki + 1) < nk;
    if (more) {  // issue next tile's global loads (overlap with WMMA below)
      const int kk = (ki + 1) << 5;
      ra0 = *(const uint4*)&A[(size_t)(m0 + row0) * K + kk + c4];
      ra1 = *(const uint4*)&A[(size_t)(m0 + row1) * K + kk + c4];
      rb0 = *(const uint4*)&W[(size_t)(n0 + row0) * K + kk + c4];
      rb1 = *(const uint4*)&W[(size_t)(n0 + row1) * K + kk + c4];
    }
    if (ki + 2 < nk)  // CDNA5 global_prefetch_b8 two tiles ahead
      __builtin_prefetch(&A[(size_t)(m0 + row0) * K + ((ki + 2) << 5) + c4],
                         0, 1);

    const v16bf a = frag_a(A_s[p], 32, wid * 16, 0, lane);
#pragma unroll
    for (int nt = 0; nt < 4; ++nt) {
      const v16bf b = frag_b(B_s[p], 32, nt * 16, 0, lane);
      acc[nt] = wmma_bf16(a, b, acc[nt]);
    }

    if (more) {
      *(uint4*)&A_s[1 - p][row0 * 32 + c4] = ra0;
      *(uint4*)&A_s[1 - p][row1 * 32 + c4] = ra1;
      *(uint4*)&B_s[1 - p][row0 * 32 + c4] = rb0;
      *(uint4*)&B_s[1 - p][row1 * 32 + c4] = rb1;
    }
    __syncthreads();
  }

  // D layout: lanes 0-15 -> M = vgpr r; lanes 16-31 -> M = 8 + r; N = lane%16
  const int half = lane >> 4;
  const int nloc = lane & 15;
#pragma unroll
  for (int nt = 0; nt < 4; ++nt) {
    const int n = n0 + nt * 16 + nloc;
    const float bv = bias ? bias[n] : 0.0f;
#pragma unroll
    for (int r = 0; r < 8; ++r) {
      const int m = m0 + wid * 16 + half * 8 + r;
      C[(size_t)m * N + n] = acc[nt][r] + bv;
    }
  }
}

// ---------------- persistent LSTM scan -------------------------------------

__device__ inline void grid_barrier(int* ctr, int target) {
  __syncthreads();
  __threadfence();
  if (threadIdx.x == 0) {
    __hip_atomic_fetch_add(ctr, 1, __ATOMIC_RELEASE, __HIP_MEMORY_SCOPE_AGENT);
    while (__hip_atomic_load(ctr, __ATOMIC_ACQUIRE,
                             __HIP_MEMORY_SCOPE_AGENT) < target)
      __builtin_amdgcn_s_sleep(2);
  }
  __syncthreads();
  __threadfence();  // invalidate L0 so fresh h is observed
}

__global__ __launch_bounds__(128) void lstm_scan(
    const float* __restrict__ xg,     // [B,T,4H] = x@W_ih^T + (b_ih+b_hh)
    const __bf16* __restrict__ Whh,   // [4H,H] bf16
    const float* __restrict__ h0,     // [B,H]
    const float* __restrict__ c0,     // [B,H]
    __bf16* __restrict__ y,           // [B,T,H] bf16 (h sequence)
    float* __restrict__ h_out,        // [B,H]
    float* __restrict__ c_out,        // [B,H]
    __bf16* __restrict__ hbuf,        // [2,B,H] bf16 double buffer
    int* __restrict__ ctr, int T) {
  __shared__ __bf16 h_s[64 * 256];    // 32KB: full h each step (TDM dest)
  __shared__ __bf16 W_s[64 * 256];    // 32KB: this WG's W_hh rows (TDM dest)

  const int tid  = threadIdx.x;
  const int lane = tid & 31;
  const int wid  = tid >> 5;
  const int wg   = blockIdx.x;
  const int j0   = wg * 16;

  const unsigned hs_off = (unsigned)(uintptr_t)(void*)h_s;
  const unsigned ws_off = (unsigned)(uintptr_t)(void*)W_s;

  // TDM preload of W_hh rows {g*256 + j0 + n : g<4, n<16}: X=256 elems,
  // Y=16 rows (stride 256), Z=4 gates (stride 256*256). TDM packs X,Y,Z
  // contiguously into LDS == W_s[(g*16+n)*256 + k].
  if (wid == 0) {
    tdm_load_3d_bf16(ws_off, Whh + (size_t)j0 * kH,
                     /*dim0=*/256, /*rows=*/16, /*planes=*/4,
                     /*stride0=*/256, /*stride1=*/256ull * 256ull,
                     /*tensor_rows=*/1024);
    __builtin_amdgcn_s_wait_tensorcnt(0);
  }

  // init h double-buffer slot 0 (this WG's 16 columns)
  for (int i = tid; i < 64 * 16; i += 128) {
    const int b = i >> 4, jj = i & 15;
    hbuf[b * kH + j0 + jj] = (__bf16)h0[b * kH + j0 + jj];
  }

  // lane-local c state: D-tile mapping b = wid*16 + 8*(lane>>4) + r
  const int half = lane >> 4;
  const int nloc = lane & 15;
  const int j    = j0 + nloc;
  float c_reg[8];
#pragma unroll
  for (int r = 0; r < 8; ++r) {
    const int b = wid * 16 + half * 8 + r;
    c_reg[r] = c0[b * kH + j];
  }

  int phase = 1;
  grid_barrier(ctr, NWG * phase);  // W_s + h init visible everywhere

  for (int t = 0; t < T; ++t) {
    const __bf16* cur = hbuf + (size_t)(t & 1) * kB * kH;
    __bf16* nxt       = hbuf + (size_t)((t + 1) & 1) * kB * kH;

    // stage full h[64,256] into LDS with one TDM 2D tile load (wave 0)
    if (wid == 0) {
      tdm_load_2d_bf16(hs_off, cur, /*dim0=*/256, /*dim1=*/64, /*stride0=*/256);
      __builtin_amdgcn_s_wait_tensorcnt(0);
    }
    __syncthreads();

    v8f ai = (v8f){0,0,0,0,0,0,0,0}, af = ai, ag = ai, ao = ai;
#pragma unroll
    for (int ks = 0; ks < 8; ++ks) {
      const int k0 = ks * 32;
      const v16bf a = frag_a(h_s, 256, wid * 16, k0, lane);
      ai = wmma_bf16(a, frag_b(W_s + 0 * 16 * 256, 256, 0, k0, lane), ai);
      af = wmma_bf16(a, frag_b(W_s + 1 * 16 * 256, 256, 0, k0, lane), af);
      ag = wmma_bf16(a, frag_b(W_s + 2 * 16 * 256, 256, 0, k0, lane), ag);
      ao = wmma_bf16(a, frag_b(W_s + 3 * 16 * 256, 256, 0, k0, lane), ao);
    }

    // lane-local LSTM cell update (torch gate order i,f,g,o)
#pragma unroll
    for (int r = 0; r < 8; ++r) {
      const int b = wid * 16 + half * 8 + r;
      const float* xr = xg + ((size_t)b * T + t) * kH4;
      const float gi = sigmoidf_(ai[r] + xr[0 * kH + j]);
      const float gf = sigmoidf_(af[r] + xr[1 * kH + j]);
      const float gg = tanhf(ag[r] + xr[2 * kH + j]);
      const float go = sigmoidf_(ao[r] + xr[3 * kH + j]);
      const float cn = gf * c_reg[r] + gi * gg;
      c_reg[r] = cn;
      const float hn = go * tanhf(cn);
      const __bf16 hb = (__bf16)hn;
      nxt[b * kH + j] = hb;
      y[((size_t)b * T + t) * kH + j] = hb;
      if (t == T - 1) h_out[b * kH + j] = hn;
      if (t + 1 < T)  // prefetch next step's gate inputs
        __builtin_prefetch(xr + kH4 + j, 0, 1);
    }

    ++phase;
    grid_barrier(ctr, NWG * phase);
  }

#pragma unroll
  for (int r = 0; r < 8; ++r) {
    const int b = wid * 16 + half * 8 + r;
    c_out[b * kH + j] = c_reg[r];
  }
}

// ---------------- host launcher --------------------------------------------

extern "C" void kernel_launch(void* const* d_in, const int* in_sizes, int n_in,
                              void* d_out, int out_size, void* d_ws,
                              size_t ws_size, hipStream_t stream) {
  const float* x    = (const float*)d_in[0];
  const float* h0   = (const float*)d_in[1];   // [2,B,H]
  const float* c0   = (const float*)d_in[2];   // [2,B,H]
  const float* Wih0 = (const float*)d_in[3];
  const float* Whh0 = (const float*)d_in[4];
  const float* bih0 = (const float*)d_in[5];
  const float* bhh0 = (const float*)d_in[6];
  const float* Wih1 = (const float*)d_in[7];
  const float* Whh1 = (const float*)d_in[8];
  const float* bih1 = (const float*)d_in[9];
  const float* bhh1 = (const float*)d_in[10];
  const float* Wout = (const float*)d_in[11];
  const float* bout = (const float*)d_in[12];

  float* out   = (float*)d_out;                       // [B,T,V]
  float* h_out = out + (size_t)kBT * kV;              // [2,B,H]
  float* c_out = h_out + (size_t)2 * kB * kH;         // [2,B,H]

  // workspace carve-out (256B aligned)
  char* ws = (char*)d_ws;
  size_t o = 0;
  auto carve = [&](size_t bytes) {
    size_t r = o;
    o += (bytes + 255) & ~(size_t)255;
    return r;
  };
  float*  xg      = (float*)(ws + carve((size_t)kBT * kH4 * 4));   // 256MB
  __bf16* x_bf    = (__bf16*)(ws + carve((size_t)kBT * kV * 2));   // 32MB
  __bf16* y0_bf   = (__bf16*)(ws + carve((size_t)kBT * kH * 2));   // 32MB
  __bf16* y1_bf   = (__bf16*)(ws + carve((size_t)kBT * kH * 2));   // 32MB
  __bf16* Wih0_bf = (__bf16*)(ws + carve((size_t)kH4 * kV * 2));
  __bf16* Whh0_bf = (__bf16*)(ws + carve((size_t)kH4 * kH * 2));
  __bf16* Wih1_bf = (__bf16*)(ws + carve((size_t)kH4 * kH * 2));
  __bf16* Whh1_bf = (__bf16*)(ws + carve((size_t)kH4 * kH * 2));
  __bf16* Wout_bf = (__bf16*)(ws + carve((size_t)kV * kH * 2));
  float*  bias0   = (float*)(ws + carve(kH4 * 4));
  float*  bias1   = (float*)(ws + carve(kH4 * 4));
  __bf16* hbuf    = (__bf16*)(ws + carve((size_t)2 * kB * kH * 2));
  int*    ctrs    = (int*)(ws + carve(256));

  hipMemsetAsync(ctrs, 0, 256, stream);

  // conversions to bf16
  cvt_f32_bf16<<<2048, 256, 0, stream>>>(x, x_bf, kBT * kV);
  cvt_f32_bf16<<<256, 256, 0, stream>>>(Wih0, Wih0_bf, kH4 * kV);
  cvt_f32_bf16<<<256, 256, 0, stream>>>(Whh0, Whh0_bf, kH4 * kH);
  cvt_f32_bf16<<<256, 256, 0, stream>>>(Wih1, Wih1_bf, kH4 * kH);
  cvt_f32_bf16<<<256, 256, 0, stream>>>(Whh1, Whh1_bf, kH4 * kH);
  cvt_f32_bf16<<<64, 256, 0, stream>>>(Wout, Wout_bf, kV * kH);
  add_vec<<<4, 256, 0, stream>>>(bih0, bhh0, bias0, kH4);
  add_vec<<<4, 256, 0, stream>>>(bih1, bhh1, bias1, kH4);

  // layer 0: xg = x_bf @ W_ih0^T + bias0 ; then scan
  {
    dim3 g(kH4 / 64, kBT / 64);
    gemm_bf16_nt<<<g, 128, 0, stream>>>(x_bf, Wih0_bf, bias0, xg,
                                        kBT, kH4, kV);
  }
  lstm_scan<<<NWG, 128, 0, stream>>>(xg, Whh0_bf, h0, c0, y0_bf,
                                     h_out, c_out, hbuf, ctrs, kT);

  // layer 1: xg = y0 @ W_ih1^T + bias1 ; then scan
  {
    dim3 g(kH4 / 64, kBT / 64);
    gemm_bf16_nt<<<g, 128, 0, stream>>>(y0_bf, Wih1_bf, bias1, xg,
                                        kBT, kH4, kH);
  }
  lstm_scan<<<NWG, 128, 0, stream>>>(xg, Whh1_bf, h0 + kB * kH, c0 + kB * kH,
                                     y1_bf, h_out + kB * kH, c_out + kB * kH,
                                     hbuf, ctrs + 64, kT);

  // output projection: out = y1 @ W_out^T + b_out
  {
    dim3 g(kV / 64, kBT / 64);
    gemm_bf16_nt<<<g, 128, 0, stream>>>(y1_bf, Wout_bf, bout, out,
                                        kBT, kV, kH);
  }
}
